// Model_74998718923011
// MI455X (gfx1250) — matrix operand order repaired and checked by reading, not compiled, not run on hardware
//
#include <hip/hip_runtime.h>
#include <hip/hip_bf16.h>

#define B_        64
#define C_IN_     3
#define T_        128
#define V_        64
#define E_        64
#define PLEN_     16
#define STRIDE_   8
#define PNUM_     15
#define N_        (PNUM_ * V_)     /* 960 */
#define FOR_LEN_  32
#define OUT_F_    3
#define NUM_LAT_  4

typedef __attribute__((ext_vector_type(16))) __bf16          v16bf;
typedef __attribute__((ext_vector_type(16))) unsigned short  v16u;
typedef __attribute__((ext_vector_type(8)))  unsigned short  v8u;
typedef __attribute__((ext_vector_type(8)))  float           v8f;
typedef int v4i __attribute__((vector_size(16)));

// ---- CDNA5 async global->LDS path (guarded; falls back to sync copy) ----
#if defined(__has_builtin)
#if __has_builtin(__builtin_amdgcn_global_load_async_to_lds_b128) && \
    __has_builtin(__builtin_amdgcn_s_wait_asynccnt)
#define HAVE_ASYNC_LDS 1
#endif
#endif

#define GLOBAL_AS __attribute__((address_space(1)))
#define LDS_AS    __attribute__((address_space(3)))

__device__ __forceinline__ void async_cp16(void* lds_dst, const void* gsrc) {
#ifdef HAVE_ASYNC_LDS
  __builtin_amdgcn_global_load_async_to_lds_b128(
      (GLOBAL_AS v4i*)(gsrc), (LDS_AS v4i*)(lds_dst), 0, 0);
#else
  *(float4*)lds_dst = *(const float4*)gsrc;
#endif
}

__device__ __forceinline__ void async_wait_all() {
#ifdef HAVE_ASYNC_LDS
  __builtin_amdgcn_s_wait_asynccnt(0);
#endif
}

__device__ __forceinline__ unsigned short f2bf(float f) {
  unsigned int u = __float_as_uint(f);
  unsigned int r = (u + 0x7FFFu + ((u >> 16) & 1u)) >> 16;  // RNE
  return (unsigned short)r;
}

__device__ __forceinline__ v16bf make_frag(v8u lo, v8u hi8) {
  v16u u = __builtin_shufflevector(lo, hi8, 0, 1, 2, 3, 4, 5, 6, 7,
                                            8, 9, 10, 11, 12, 13, 14, 15);
  return __builtin_bit_cast(v16bf, u);
}

// A fragment (16x32 bf16): LDS row-major As[row][k], 64 halves/row, 16B-aligned.
__device__ __forceinline__ v16bf load_a_frag(const unsigned short* As,
                                             int row, int ks, int lane) {
  const int hi = (lane >= 16) ? 1 : 0;
  const v8u* p = (const v8u*)(As + row * 64 + ks);
  return make_frag(p[hi], p[2 + hi]);
}

// B fragment (32x16 bf16): LDS stored TRANSPOSED Bt[n][k] (K contiguous).
__device__ __forceinline__ v16bf load_bt_frag(const unsigned short* Bt,
                                              int ks, int n0, int lane, int ldk) {
  const int hi = (lane >= 16) ? 1 : 0;
  const int n = n0 + (lane & 15);
  const v8u* p = (const v8u*)(Bt + n * ldk + ks);
  return make_frag(p[hi], p[2 + hi]);
}

__device__ __forceinline__ v8f wmma_bf16(v16bf a, v16bf b, v8f c) {
  return __builtin_amdgcn_wmma_f32_16x16x32_bf16(false, a, false, b,
                                                 (short)0, c, false, false);
}

// ---------------- 1) |FFT| along T via twiddle-table DFT ----------------
__global__ void k_fft_abs(const float* __restrict__ x, float* __restrict__ out) {
  __shared__ float xcol[T_];
  __shared__ float ctab[T_];
  __shared__ float stab[T_];
  const int bc = blockIdx.x;            // b*C_IN + c
  const int t  = threadIdx.x;           // 0..127
  const float w = 6.283185307179586f / (float)T_;
  ctab[t] = __cosf(w * (float)t);
  stab[t] = __sinf(w * (float)t);
  const float* xb = x + (size_t)bc * T_ * V_;
  float* ob       = out + (size_t)bc * T_ * V_;
  for (int v = 0; v < V_; ++v) {
    __syncthreads();
    xcol[t] = xb[t * V_ + v];
    __syncthreads();
    float re = 0.f, im = 0.f;
    for (int tau = 0; tau < T_; ++tau) {
      int k = (t * tau) & (T_ - 1);
      float xv = xcol[tau];
      re = fmaf(xv, ctab[k], re);
      im = fmaf(xv, stab[k], im);
    }
    ob[t * V_ + v] = sqrtf(re * re + im * im);
  }
}

// ---------------- 2) patch embed: gather -> WMMA (K 48 padded to 64) ----
__global__ void k_patch_embed(const float* __restrict__ src, const float* __restrict__ W,
                              const float* __restrict__ bias, float* __restrict__ out) {
  alignas(16) __shared__ unsigned short As[64 * 64];   // [row][k]
  alignas(16) __shared__ unsigned short Bt[64 * 64];   // [n][k] (transposed W)
  const int blk = blockIdx.x;                // B_*PNUM_
  const int b = blk / PNUM_, r0 = (blk % PNUM_) * 64;
  const int tid = threadIdx.x, lane = tid & 31, wv = tid >> 5;
  const float* sb = src + (size_t)b * C_IN_ * T_ * V_;
  for (int c = tid; c < 512; c += 128) {
    int rr = c >> 3, k8 = (c & 7) * 8;
    int r = r0 + rr;
    int v = r / PNUM_, p = r % PNUM_;
    v8u u;
#pragma unroll
    for (int j = 0; j < 8; ++j) {
      int k = k8 + j;
      float val = 0.f;
      if (k < C_IN_ * PLEN_) {
        int cc = k >> 4, l = k & 15;
        val = sb[((size_t)cc * T_ + (p * STRIDE_ + l)) * V_ + v];
      }
      u[j] = f2bf(val);
    }
    ((v8u*)As)[c] = u;
  }
  for (int c = tid; c < 512; c += 128) {
    int n = c >> 3, k8 = (c & 7) * 8;
    v8u u;
#pragma unroll
    for (int j = 0; j < 8; ++j) {
      int k = k8 + j;
      u[j] = (k < C_IN_ * PLEN_) ? f2bf(W[k * E_ + n]) : (unsigned short)0;
    }
    ((v8u*)Bt)[c] = u;
  }
  __syncthreads();
  const int arow = wv * 16 + (lane & 15);
  for (int nt = 0; nt < 4; ++nt) {
    v8f acc = {0.f, 0.f, 0.f, 0.f, 0.f, 0.f, 0.f, 0.f};
#pragma unroll
    for (int ks = 0; ks < 64; ks += 32) {
      v16bf a = load_a_frag(As, arow, ks, lane);
      v16bf bf = load_bt_frag(Bt, ks, nt * 16, lane, 64);
      acc = wmma_bf16(a, bf, acc);
    }
    const int n = nt * 16 + (lane & 15);
    const int rbase = wv * 16 + ((lane >= 16) ? 8 : 0);
#pragma unroll
    for (int r = 0; r < 8; ++r)
      out[((size_t)b * N_ + r0 + rbase + r) * E_ + n] = acc[r] + bias[n];
  }
}

// ---------------- 3a) column mean over n (per batch) --------------------
__global__ void k_mean(const float* __restrict__ x, float* __restrict__ mean) {
  const int b = blockIdx.x, e = threadIdx.x;
  const float* xb = x + (size_t)b * N_ * E_ + e;
  float s = 0.f;
  for (int n = 0; n < N_; ++n) s += xb[(size_t)n * E_];
  mean[b * E_ + e] = s * (1.0f / (float)N_);
}

// 3b) cheb(x) = x @ (W0-W2) + mean @ (2W2-W1) + bias   [rank-1 Laplacian]
// A-tile staged via GLOBAL_LOAD_ASYNC_TO_LDS (raw f32), then packed to bf16.
__global__ void k_cheb(float* __restrict__ x, const float* __restrict__ mean,
                       const float* __restrict__ W, const float* __restrict__ bias,
                       int residual) {
  alignas(16) __shared__ float          Xf[64 * 64];   // raw f32 tile (async dest)
  alignas(16) __shared__ unsigned short As[64 * 64];   // [row][k] bf16
  alignas(16) __shared__ unsigned short Wt[64 * 64];   // [n][k] = (W0-W2)^T
  __shared__ float mrow[64];
  const int blk = blockIdx.x;
  const int b = blk / PNUM_, n0 = (blk % PNUM_) * 64;
  const int tid = threadIdx.x, lane = tid & 31, wv = tid >> 5;
  const float* W0 = W;
  const float* W1 = W + E_ * E_;
  const float* W2 = W + 2 * E_ * E_;
  float* xb = x + (size_t)b * N_ * E_;
  // async DMA: contiguous 16KB tile, 1024 x 16B per-lane copies
  const float* tile = xb + (size_t)n0 * E_;
  for (int c = tid; c < 1024; c += 128)
    async_cp16(Xf + c * 4, tile + c * 4);
  // weights (transposed) via normal path
  for (int c = tid; c < 512; c += 128) {
    int n = c >> 3, k8 = (c & 7) * 8;
    v8u u;
#pragma unroll
    for (int j = 0; j < 8; ++j) {
      int k = k8 + j;
      u[j] = f2bf(W0[k * E_ + n] - W2[k * E_ + n]);
    }
    ((v8u*)Wt)[c] = u;
  }
  if (tid < 64) {
    float acc = 0.f;
    for (int k = 0; k < 64; ++k)
      acc = fmaf(mean[b * E_ + k], 2.f * W2[k * E_ + tid] - W1[k * E_ + tid], acc);
    mrow[tid] = acc + bias[tid];
  }
  async_wait_all();
  __syncthreads();                       // Xf, Wt, mrow ready
  // pack f32 -> bf16 (LDS-local)
  for (int c = tid; c < 512; c += 128) {
    const float4* s = (const float4*)(Xf + c * 8);
    float4 f0 = s[0], f1 = s[1];
    v8u u;
    u[0] = f2bf(f0.x); u[1] = f2bf(f0.y); u[2] = f2bf(f0.z); u[3] = f2bf(f0.w);
    u[4] = f2bf(f1.x); u[5] = f2bf(f1.y); u[6] = f2bf(f1.z); u[7] = f2bf(f1.w);
    ((v8u*)As)[c] = u;
  }
  __syncthreads();                       // As ready
  const int arow = wv * 16 + (lane & 15);
  for (int nt = 0; nt < 4; ++nt) {
    v8f acc = {0.f, 0.f, 0.f, 0.f, 0.f, 0.f, 0.f, 0.f};
#pragma unroll
    for (int ks = 0; ks < 64; ks += 32) {
      v16bf a = load_a_frag(As, arow, ks, lane);
      v16bf bf = load_bt_frag(Wt, ks, nt * 16, lane, 64);
      acc = wmma_bf16(a, bf, acc);
    }
    const int n = nt * 16 + (lane & 15);
    const int rbase = wv * 16 + ((lane >= 16) ? 8 : 0);
#pragma unroll
    for (int r = 0; r < 8; ++r) {
      const int rl = rbase + r;          // 0..63 within tile
      float val = acc[r] + mrow[n];
      if (residual) val += Xf[rl * 64 + n];   // residual from LDS stage
      xb[(size_t)(n0 + rl) * E_ + n] = val;   // in-place: rows owned by block
    }
  }
}

// ---------------- 4a) latent attention: fl = attn(lat, concat, concat) --
__global__ void k_latent_attn(const float* __restrict__ time, const float* __restrict__ fre,
                              const float* __restrict__ lat, float* __restrict__ fl) {
  __shared__ float sc[NUM_LAT_][2 * N_];
  __shared__ float inv[NUM_LAT_];
  const int b = blockIdx.x, tid = threadIdx.x;   // 256
  const float* tb = time + (size_t)b * N_ * E_;
  const float* fb = fre + (size_t)b * N_ * E_;
  for (int i = tid; i < NUM_LAT_ * 2 * N_; i += 256) {
    int j = i / (2 * N_), s = i % (2 * N_);
    const float* row = (s < N_) ? (tb + (size_t)s * E_) : (fb + (size_t)(s - N_) * E_);
    const float* q = lat + j * E_;
    float d = 0.f;
    for (int e = 0; e < E_; ++e) d = fmaf(row[e], q[e], d);
    sc[j][s] = d * 0.125f;                        // E^-0.5
  }
  __syncthreads();
  if (tid < NUM_LAT_) {
    float m = -3.4e38f;
    for (int s = 0; s < 2 * N_; ++s) m = fmaxf(m, sc[tid][s]);
    float sum = 0.f;
    for (int s = 0; s < 2 * N_; ++s) { float e = __expf(sc[tid][s] - m); sc[tid][s] = e; sum += e; }
    inv[tid] = 1.f / sum;
  }
  __syncthreads();
  const int j = tid >> 6, e = tid & 63;
  float acc = 0.f;
  for (int s = 0; s < 2 * N_; ++s) {
    const float* row = (s < N_) ? (tb + (size_t)s * E_) : (fb + (size_t)(s - N_) * E_);
    acc = fmaf(sc[j][s], row[e], acc);
  }
  fl[((size_t)b * NUM_LAT_ + j) * E_ + e] = acc * inv[j];
}

// 4b) x += scale * attn(x, fl, fl)   (softmax over 4 latents, per row)
__global__ void k_cross_attn(float* __restrict__ x, const float* __restrict__ fl,
                             const float* __restrict__ scale_arr, int si) {
  __shared__ float fls[NUM_LAT_][E_];
  const int blk = blockIdx.x;
  const int b = blk / PNUM_, n0 = (blk % PNUM_) * 64;
  const int tid = threadIdx.x;                    // 64
  for (int i = tid; i < NUM_LAT_ * E_; i += 64)
    fls[i >> 6][i & 63] = fl[(size_t)b * NUM_LAT_ * E_ + i];
  __syncthreads();
  const float scale = scale_arr[si];
  float* row = x + ((size_t)b * N_ + n0 + tid) * E_;
  float s[NUM_LAT_] = {0.f, 0.f, 0.f, 0.f};
  for (int e = 0; e < E_; ++e) {
    float xv = row[e];
#pragma unroll
    for (int j = 0; j < NUM_LAT_; ++j) s[j] = fmaf(xv, fls[j][e], s[j]);
  }
  float m = -3.4e38f;
#pragma unroll
  for (int j = 0; j < NUM_LAT_; ++j) { s[j] *= 0.125f; m = fmaxf(m, s[j]); }
  float sum = 0.f;
#pragma unroll
  for (int j = 0; j < NUM_LAT_; ++j) { s[j] = __expf(s[j] - m); sum += s[j]; }
  const float inv = scale / sum;
  for (int e = 0; e < E_; ++e) {
    float y = 0.f;
#pragma unroll
    for (int j = 0; j < NUM_LAT_; ++j) y = fmaf(s[j], fls[j][e], y);
    row[e] += inv * y;
  }
}

// ---------------- 5) decoder: (B*V)x960x96 WMMA GEMM + transpose -------
// time/fre rows async-staged to LDS f32, fused 0.5*(t+f) -> bf16 pack.
__global__ void k_decoder(const float* __restrict__ time, const float* __restrict__ fre,
                          const float* __restrict__ dw, const float* __restrict__ db,
                          float* __restrict__ out) {
  alignas(16) __shared__ float          Tf[64 * 64];   // async dest (time rows)
  alignas(16) __shared__ float          Ff[64 * 64];   // async dest (fre rows)
  alignas(16) __shared__ unsigned short As[64 * 64];   // [v][e] bf16
  alignas(16) __shared__ unsigned short Bt[96 * 64];   // [n][k] (transposed dec_w)
  const int b = blockIdx.x, tid = threadIdx.x, lane = tid & 31, wv = tid >> 5;
  const float* tb = time + (size_t)b * N_ * E_;
  const float* fb = fre + (size_t)b * N_ * E_;
  v8f zero = {0.f, 0.f, 0.f, 0.f, 0.f, 0.f, 0.f, 0.f};
  v8f acc[6];
#pragma unroll
  for (int nt = 0; nt < 6; ++nt) acc[nt] = zero;
  const int arow = wv * 16 + (lane & 15);
  for (int kc = 0; kc < PNUM_; ++kc) {
    __syncthreads();                                   // protect prior-iter reads
    // async DMA: 64 rows x 256B from each of time/fre (per-lane 16B copies)
    for (int c = tid; c < 1024; c += 128) {
      int v = c >> 4, part = (c & 15) * 4;
      size_t idx = (size_t)(v * PNUM_ + kc) * E_ + part;
      async_cp16(Tf + c * 4, tb + idx);
      async_cp16(Ff + c * 4, fb + idx);
    }
    // Bt[n][k] = dec_w[kc*64+k][n]
    for (int c = tid; c < 768; c += 128) {
      int n = c >> 3, k8 = (c & 7) * 8;
      v8u u;
#pragma unroll
      for (int j = 0; j < 8; ++j)
        u[j] = f2bf(dw[(size_t)(kc * 64 + k8 + j) * 96 + n]);
      ((v8u*)Bt)[c] = u;
    }
    if (kc + 1 < PNUM_)
      __builtin_prefetch(&dw[(size_t)((kc + 1) * 64) * 96], 0, 1);
    async_wait_all();
    __syncthreads();                                   // Tf/Ff/Bt ready
    for (int c = tid; c < 512; c += 128) {             // fuse + pack to bf16
      const float4* pt = (const float4*)(Tf + c * 8);
      const float4* pf = (const float4*)(Ff + c * 8);
      float4 t0 = pt[0], t1 = pt[1], f0 = pf[0], f1 = pf[1];
      v8u u;
      u[0] = f2bf(0.5f * (t0.x + f0.x)); u[1] = f2bf(0.5f * (t0.y + f0.y));
      u[2] = f2bf(0.5f * (t0.z + f0.z)); u[3] = f2bf(0.5f * (t0.w + f0.w));
      u[4] = f2bf(0.5f * (t1.x + f1.x)); u[5] = f2bf(0.5f * (t1.y + f1.y));
      u[6] = f2bf(0.5f * (t1.z + f1.z)); u[7] = f2bf(0.5f * (t1.w + f1.w));
      ((v8u*)As)[c] = u;
    }
    __syncthreads();                                   // As ready
#pragma unroll
    for (int nt = 0; nt < 6; ++nt) {
#pragma unroll
      for (int ks = 0; ks < 64; ks += 32) {
        v16bf a = load_a_frag(As, arow, ks, lane);
        v16bf bf = load_bt_frag(Bt, ks, nt * 16, lane, 64);
        acc[nt] = wmma_bf16(a, bf, acc[nt]);
      }
    }
  }
  const int rbase = wv * 16 + ((lane >= 16) ? 8 : 0);
#pragma unroll
  for (int nt = 0; nt < 6; ++nt) {
    const int n = nt * 16 + (lane & 15);                // n = l*OUT_F + f
    const int l = n / OUT_F_, f = n % OUT_F_;
#pragma unroll
    for (int r = 0; r < 8; ++r) {
      const int v = rbase + r;
      out[(((size_t)b * OUT_F_ + f) * FOR_LEN_ + l) * V_ + v] = acc[nt][r] + db[n];
    }
  }
}

extern "C" void kernel_launch(void* const* d_in, const int* in_sizes, int n_in,
                              void* d_out, int out_size, void* d_ws, size_t ws_size,
                              hipStream_t stream) {
  const float* x        = (const float*)d_in[0];
  const float* w_temb   = (const float*)d_in[1];
  const float* b_temb   = (const float*)d_in[2];
  const float* w_femb   = (const float*)d_in[3];
  const float* b_femb   = (const float*)d_in[4];
  const float* t_enc_w  = (const float*)d_in[5];
  const float* t_enc_b  = (const float*)d_in[6];
  const float* f_enc_w  = (const float*)d_in[7];
  const float* f_enc_b  = (const float*)d_in[8];
  const float* fus_tg_w = (const float*)d_in[9];
  const float* fus_tg_b = (const float*)d_in[10];
  const float* fus_fg_w = (const float*)d_in[11];
  const float* fus_fg_b = (const float*)d_in[12];
  const float* fus_lat  = (const float*)d_in[13];
  const float* fus_st   = (const float*)d_in[14];
  const float* fus_sf   = (const float*)d_in[15];
  const float* dec_w    = (const float*)d_in[16];
  const float* dec_b    = (const float*)d_in[17];
  float* out = (float*)d_out;

  float* ws     = (float*)d_ws;
  float* fre_in = ws;                                       // B*C_IN*T*V
  float* timeb  = fre_in + (size_t)B_ * C_IN_ * T_ * V_;    // B*N*E
  float* freb   = timeb + (size_t)B_ * N_ * E_;             // B*N*E
  float* meanb  = freb + (size_t)B_ * N_ * E_;              // B*E
  float* flb    = meanb + (size_t)B_ * E_;                  // B*4*E

  k_fft_abs<<<B_ * C_IN_, T_, 0, stream>>>(x, fre_in);
  k_patch_embed<<<B_ * PNUM_, 128, 0, stream>>>(x, w_temb, b_temb, timeb);
  k_patch_embed<<<B_ * PNUM_, 128, 0, stream>>>(fre_in, w_femb, b_femb, freb);

  for (int i = 0; i < 2; ++i) {
    k_mean<<<B_, E_, 0, stream>>>(freb, meanb);
    k_cheb<<<B_ * PNUM_, 128, 0, stream>>>(freb, meanb,
        f_enc_w + (size_t)i * 3 * E_ * E_, f_enc_b + i * E_, 0);
    k_mean<<<B_, E_, 0, stream>>>(timeb, meanb);
    k_cheb<<<B_ * PNUM_, 128, 0, stream>>>(timeb, meanb,
        t_enc_w + (size_t)i * 3 * E_ * E_, t_enc_b + i * E_, 0);
  }
  for (int i = 0; i < 2; ++i) {
    k_latent_attn<<<B_, 256, 0, stream>>>(timeb, freb,
        fus_lat + (size_t)i * NUM_LAT_ * E_, flb);
    k_cross_attn<<<B_ * PNUM_, 64, 0, stream>>>(timeb, flb, fus_st, i);
    k_cross_attn<<<B_ * PNUM_, 64, 0, stream>>>(freb, flb, fus_sf, i);
    k_mean<<<B_, E_, 0, stream>>>(timeb, meanb);
    k_cheb<<<B_ * PNUM_, 128, 0, stream>>>(timeb, meanb,
        fus_tg_w + (size_t)i * 3 * E_ * E_, fus_tg_b + i * E_, 1);
    k_mean<<<B_, E_, 0, stream>>>(freb, meanb);
    k_cheb<<<B_ * PNUM_, 128, 0, stream>>>(freb, meanb,
        fus_fg_w + (size_t)i * 3 * E_ * E_, fus_fg_b + i * E_, 1);
  }
  k_decoder<<<B_, 128, 0, stream>>>(timeb, freb, dec_w, dec_b, out);
}
